// MaskedMultiHeadAttention_77017353552211
// MI455X (gfx1250) — compile-verified, matrix operand order checked by reading, not run on hardware
//
#include <hip/hip_runtime.h>

// Problem constants (reference: B=4, S=2048, D=1024, H=16, DH=64)
#define Bb  4
#define Ss  2048
#define Dd  1024
#define Hh  16
#define DHh 64

typedef __attribute__((ext_vector_type(16))) __bf16 v16bf;
typedef __attribute__((ext_vector_type(8)))  float  v8f;
typedef unsigned short u16;
typedef unsigned int   u32;

// ---------------------------------------------------------------------------
// helpers
// ---------------------------------------------------------------------------
__device__ __forceinline__ u16 f2bf(float f) {
  union { float f; u32 u; } c; c.f = f;
  u32 u = c.u;
  u += 0x7FFFu + ((u >> 16) & 1u);      // round-to-nearest-even
  return (u16)(u >> 16);
}
__device__ __forceinline__ u32 pack2(float a, float b) {
  return (u32)f2bf(a) | ((u32)f2bf(b) << 16);
}

union FragU { v16bf v; u32 u[8]; };

// Build a 16x32 bf16 WMMA A/B fragment for this lane from a K-major row of
// bf16 data. Per the CDNA5 ISA 16-bit A layout, element pair p of lane-half
// `half` holds K = (p&4 ? 16 : 0) + 8*half + 2*(p&3) .. +1 (pairs contiguous
// in K -> one aligned 32-bit load per pair).
__device__ __forceinline__ v16bf frag_row(const u16* row, int half) {
  FragU f;
#pragma unroll
  for (int p = 0; p < 8; ++p) {
    int k = ((p & 4) ? 16 : 0) + 8 * half + ((p & 3) << 1);
    f.u[p] = *(const u32*)(row + k);
  }
  return f.v;
}

__device__ __forceinline__ v8f wmma_bf16(v16bf a, v16bf b, v8f c) {
  return __builtin_amdgcn_wmma_f32_16x16x32_bf16(false, a, false, b,
                                                 (short)0, c, false, false);
}

// LDS row padding (elements): keeps 4B alignment (even) and spreads banks.
#define LDSP 34
#define KPAD 68

// ---------------------------------------------------------------------------
// Kernel 1: QKV projection. Y = X @ W^T + b  (both operands K-major rows),
// output bf16 scattered to [B, H, S, DH]. grid = (64, 8, 3); block = 256.
// Block tile 128x128, wave tile 64x32, Kc = 32.
// ---------------------------------------------------------------------------
__global__ __launch_bounds__(256) void qkv_proj_kernel(
    const float* __restrict__ x,
    const float* __restrict__ Wq, const float* __restrict__ Wk,
    const float* __restrict__ Wv,
    const float* __restrict__ bq, const float* __restrict__ bk,
    const float* __restrict__ bv,
    u16* __restrict__ Qo, u16* __restrict__ Ko, u16* __restrict__ Vo) {
  __shared__ u16 As[128 * LDSP];
  __shared__ u16 Bs[128 * LDSP];

  const int z = blockIdx.z;
  const float* W    = (z == 0) ? Wq : (z == 1) ? Wk : Wv;
  const float* bias = (z == 0) ? bq : (z == 1) ? bk : bv;
  u16* out          = (z == 0) ? Qo : (z == 1) ? Ko : Vo;

  const int mBase = blockIdx.x * 128;
  const int nBase = blockIdx.y * 128;
  const int tid  = threadIdx.x;
  const int wave = tid >> 5, lane = tid & 31;
  const int half = lane >> 4, lr = lane & 15;
  const int mW = (wave >> 2) * 64;    // 0 / 64
  const int nW = (wave & 3) * 32;     // 0,32,64,96

  v8f acc[4][2] = {};

  for (int kb = 0; kb < Dd; kb += 32) {
    __syncthreads();
    // Stage X chunk (128x32 f32->bf16) and W chunk (128x32 f32->bf16).
#pragma unroll
    for (int r = 0; r < 4; ++r) {
      int v  = tid + r * 256;          // 0..1023 float4 slots
      int m  = v >> 3;
      int kq = (v & 7) << 2;
      float4 xv = *(const float4*)(x + (size_t)(mBase + m) * Dd + kb + kq);
      *(u32*)(As + m * LDSP + kq)     = pack2(xv.x, xv.y);
      *(u32*)(As + m * LDSP + kq + 2) = pack2(xv.z, xv.w);
      float4 wv = *(const float4*)(W + (size_t)(nBase + m) * Dd + kb + kq);
      *(u32*)(Bs + m * LDSP + kq)     = pack2(wv.x, wv.y);
      *(u32*)(Bs + m * LDSP + kq + 2) = pack2(wv.z, wv.w);
    }
    __syncthreads();

    v16bf af[4], bf[2];
#pragma unroll
    for (int i = 0; i < 4; ++i)
      af[i] = frag_row(As + (mW + i * 16 + lr) * LDSP, half);
#pragma unroll
    for (int j = 0; j < 2; ++j)
      bf[j] = frag_row(Bs + (nW + j * 16 + lr) * LDSP, half);
#pragma unroll
    for (int i = 0; i < 4; ++i)
#pragma unroll
      for (int j = 0; j < 2; ++j)
        acc[i][j] = wmma_bf16(af[i], bf[j], acc[i][j]);
  }

  // Epilogue: + bias, f32->bf16, scatter to [B, H, S, DH].
#pragma unroll
  for (int j = 0; j < 2; ++j) {
    int n = nBase + nW + j * 16 + lr;
    float bj = bias[n];
    int h = n >> 6, dh = n & 63;
#pragma unroll
    for (int i = 0; i < 4; ++i) {
      int mRow = mBase + mW + i * 16 + 8 * half;
#pragma unroll
      for (int e = 0; e < 8; ++e) {
        int m = mRow + e;
        int b = m >> 11, s = m & 2047;
        out[(((size_t)b * Hh + h) * Ss + s) * DHh + dh] = f2bf(acc[i][j][e] + bj);
      }
    }
  }
}

// ---------------------------------------------------------------------------
// Kernel 2: flash attention, causal. One block per (b, h, 128 q rows);
// 8 waves x 16 q rows. K/V tiles (64 kpos) staged in LDS, V transposed.
// grid = (B*H*S/128) = 1024; block = 256.
// ---------------------------------------------------------------------------
__global__ __launch_bounds__(256) void attn_kernel(
    const u16* __restrict__ Q, const u16* __restrict__ K,
    const u16* __restrict__ V, u16* __restrict__ Ctx) {
  __shared__ u16 Ks[64 * KPAD];           // [kpos][d]
  __shared__ u16 Vs[64 * KPAD];           // transposed: [dh][kpos]
  __shared__ u16 Ps[8 * 16 * KPAD];       // per-wave P tile [qrow][kpos]

  const int qChunk = blockIdx.x & 15;
  const int bh = blockIdx.x >> 4;          // b*H + h
  const int b = bh >> 4, h = bh & 15;
  const int tid  = threadIdx.x;
  const int wave = tid >> 5, lane = tid & 31;
  const int half = lane >> 4, lr = lane & 15;
  const int qBase = qChunk * 128 + wave * 16;

  const u16* Qbh = Q + (size_t)bh * Ss * DHh;
  const u16* Kbh = K + (size_t)bh * Ss * DHh;
  const u16* Vbh = V + (size_t)bh * Ss * DHh;
  u16* Pw = Ps + wave * 16 * KPAD;

  // Q fragments: 16 rows x DH=64 -> two 16x32 A frags, loaded once.
  v16bf qf[2];
#pragma unroll
  for (int c = 0; c < 2; ++c)
    qf[c] = frag_row(Qbh + (size_t)(qBase + lr) * DHh + c * 32, half);

  float mrow[8], lsum[8];
  v8f cacc[4] = {};
#pragma unroll
  for (int e = 0; e < 8; ++e) { mrow[e] = -1e30f; lsum[e] = 0.f; }

  const int maxKT = (qChunk * 128 + 127) >> 6;
  for (int kt = 0; kt <= maxKT; ++kt) {
    __syncthreads();
    // Stage K tile (row-major copy, 32-bit) and V tile (transposed).
#pragma unroll
    for (int r = 0; r < 8; ++r) {
      int v  = tid + r * 256;              // 0..2047 u32 slots
      int kp = v >> 5;
      int d2 = (v & 31) << 1;
      *(u32*)(Ks + kp * KPAD + d2) =
          *(const u32*)(Kbh + (size_t)(kt * 64 + kp) * DHh + d2);
    }
#pragma unroll
    for (int r = 0; r < 16; ++r) {
      int v  = tid + r * 256;              // 0..4095 elements
      int kp = v >> 6;
      int dh = v & 63;
      Vs[dh * KPAD + kp] = Vbh[(size_t)(kt * 64 + kp) * DHh + dh];
    }
    __syncthreads();

    if (kt * 64 <= qBase + 15) {           // wave-uniform: EXEC stays all-ones
      // scores S = Q @ K^T : 16 x 64
      v8f sacc[4] = {};
#pragma unroll
      for (int c = 0; c < 2; ++c)
#pragma unroll
        for (int t = 0; t < 4; ++t) {
          v16bf kf = frag_row(Ks + (t * 16 + lr) * KPAD + c * 32, half);
          sacc[t] = wmma_bf16(qf[c], kf, sacc[t]);
        }

      // scale 1/sqrt(DH) + causal mask (C layout: row = e + 8*half, col = lr)
      float sc[4][8];
#pragma unroll
      for (int t = 0; t < 4; ++t) {
        int kp = kt * 64 + t * 16 + lr;
#pragma unroll
        for (int e = 0; e < 8; ++e) {
          int q = qBase + 8 * half + e;
          float s = sacc[t][e] * 0.125f;
          sc[t][e] = (kp > q) ? -1e30f : s;
        }
      }

      // online softmax: per-row max across 64 cols (4 accs x 16 lanes/half)
      float mnew[8], corr[8], psum[8];
#pragma unroll
      for (int e = 0; e < 8; ++e) {
        float mx = fmaxf(fmaxf(sc[0][e], sc[1][e]), fmaxf(sc[2][e], sc[3][e]));
#pragma unroll
        for (int off = 1; off < 16; off <<= 1)
          mx = fmaxf(mx, __shfl_xor(mx, off, 32));
        mnew[e] = fmaxf(mrow[e], mx);
        corr[e] = __expf(mrow[e] - mnew[e]);
        mrow[e] = mnew[e];
        lsum[e] *= corr[e];
        psum[e] = 0.f;
      }
      // exp, accumulate row sums, spill P (bf16) to per-wave LDS
#pragma unroll
      for (int t = 0; t < 4; ++t)
#pragma unroll
        for (int e = 0; e < 8; ++e) {
          float p = __expf(sc[t][e] - mnew[e]);
          psum[e] += p;
          Pw[(8 * half + e) * KPAD + t * 16 + lr] = f2bf(p);
        }
#pragma unroll
      for (int t = 0; t < 4; ++t)
#pragma unroll
        for (int e = 0; e < 8; ++e) cacc[t][e] *= corr[e];
#pragma unroll
      for (int e = 0; e < 8; ++e) {
#pragma unroll
        for (int off = 1; off < 16; off <<= 1)
          psum[e] += __shfl_xor(psum[e], off, 32);
        lsum[e] += psum[e];
      }

      // cross-lane LDS dependency (C-layout store -> A-layout load):
      asm volatile("s_wait_dscnt 0" ::: "memory");

      // ctx += P @ V   (A = P from LDS, B = transposed V tile)
#pragma unroll
      for (int c = 0; c < 2; ++c) {
        v16bf pf = frag_row(Pw + lr * KPAD + c * 32, half);
#pragma unroll
        for (int t = 0; t < 4; ++t) {
          v16bf vf = frag_row(Vs + (t * 16 + lr) * KPAD + c * 32, half);
          cacc[t] = wmma_bf16(pf, vf, cacc[t]);
        }
      }
    }
  }

  // Normalize and store ctx (bf16) as [B, S, H, DH] == [8192 x 1024] matrix.
#pragma unroll
  for (int e = 0; e < 8; ++e) {
    float inv = 1.0f / lsum[e];
    int q = qBase + 8 * half + e;
#pragma unroll
    for (int t = 0; t < 4; ++t) {
      int dh = t * 16 + lr;
      Ctx[(((size_t)b * Ss + q) * Hh + h) * DHh + dh] = f2bf(cacc[t][e] * inv);
    }
  }
}

// ---------------------------------------------------------------------------
// Kernel 3: output projection. out = Ctx @ Wo^T + bo, fp32 output.
// grid = (64, 8); block = 256. Same tiling as kernel 1, A already bf16.
// ---------------------------------------------------------------------------
__global__ __launch_bounds__(256) void out_proj_kernel(
    const u16* __restrict__ ctx, const float* __restrict__ Wo,
    const float* __restrict__ bo, float* __restrict__ out) {
  __shared__ u16 As[128 * LDSP];
  __shared__ u16 Bs[128 * LDSP];

  const int mBase = blockIdx.x * 128;
  const int nBase = blockIdx.y * 128;
  const int tid  = threadIdx.x;
  const int wave = tid >> 5, lane = tid & 31;
  const int half = lane >> 4, lr = lane & 15;
  const int mW = (wave >> 2) * 64;
  const int nW = (wave & 3) * 32;

  v8f acc[4][2] = {};

  for (int kb = 0; kb < Dd; kb += 32) {
    __syncthreads();
#pragma unroll
    for (int r = 0; r < 8; ++r) {          // ctx chunk: straight bf16 copy
      int v  = tid + r * 256;              // 0..2047 u32 slots
      int m  = v >> 4;
      int kq = (v & 15) << 1;
      *(u32*)(As + m * LDSP + kq) =
          *(const u32*)(ctx + (size_t)(mBase + m) * Dd + kb + kq);
    }
#pragma unroll
    for (int r = 0; r < 4; ++r) {          // Wo chunk: f32 -> bf16
      int v  = tid + r * 256;
      int m  = v >> 3;
      int kq = (v & 7) << 2;
      float4 wv = *(const float4*)(Wo + (size_t)(nBase + m) * Dd + kb + kq);
      *(u32*)(Bs + m * LDSP + kq)     = pack2(wv.x, wv.y);
      *(u32*)(Bs + m * LDSP + kq + 2) = pack2(wv.z, wv.w);
    }
    __syncthreads();

    v16bf af[4], bf[2];
#pragma unroll
    for (int i = 0; i < 4; ++i)
      af[i] = frag_row(As + (mW + i * 16 + lr) * LDSP, half);
#pragma unroll
    for (int j = 0; j < 2; ++j)
      bf[j] = frag_row(Bs + (nW + j * 16 + lr) * LDSP, half);
#pragma unroll
    for (int i = 0; i < 4; ++i)
#pragma unroll
      for (int j = 0; j < 2; ++j)
        acc[i][j] = wmma_bf16(af[i], bf[j], acc[i][j]);
  }

#pragma unroll
  for (int j = 0; j < 2; ++j) {
    int n = nBase + nW + j * 16 + lr;
    float bj = bo[n];
#pragma unroll
    for (int i = 0; i < 4; ++i) {
      int mRow = mBase + mW + i * 16 + 8 * half;
#pragma unroll
      for (int e = 0; e < 8; ++e)
        out[(size_t)(mRow + e) * Dd + n] = acc[i][j][e] + bj;
    }
  }
}

// ---------------------------------------------------------------------------
// launch
// ---------------------------------------------------------------------------
extern "C" void kernel_launch(void* const* d_in, const int* in_sizes, int n_in,
                              void* d_out, int out_size, void* d_ws,
                              size_t ws_size, hipStream_t stream) {
  (void)in_sizes; (void)n_in; (void)out_size; (void)ws_size;

  const float* x  = (const float*)d_in[0];
  const float* Wq = (const float*)d_in[1];
  const float* bq = (const float*)d_in[2];
  const float* Wk = (const float*)d_in[3];
  const float* bk = (const float*)d_in[4];
  const float* Wv = (const float*)d_in[5];
  const float* bv = (const float*)d_in[6];
  const float* Wo = (const float*)d_in[7];
  const float* bo = (const float*)d_in[8];
  float* out = (float*)d_out;

  // Workspace: Q, K, V as bf16 [B,H,S,DH]; Ctx as bf16 [B,S,H,DH]. 64 MB.
  const size_t elems = (size_t)Bb * Ss * Dd;   // 8M
  u16* Qw = (u16*)d_ws;
  u16* Kw = Qw + elems;
  u16* Vw = Kw + elems;
  u16* Cw = Vw + elems;

  qkv_proj_kernel<<<dim3(64, 8, 3), 256, 0, stream>>>(
      x, Wq, Wk, Wv, bq, bk, bv, Qw, Kw, Vw);
  attn_kernel<<<dim3((Bb * Hh) * (Ss / 128)), 256, 0, stream>>>(Qw, Kw, Vw, Cw);
  out_proj_kernel<<<dim3(64, 8), 256, 0, stream>>>(Cw, Wo, bo, out);
}
